// AttentionModel_64175401337266
// MI455X (gfx1250) — matrix operand order repaired and checked by reading, not compiled, not run on hardware
//
#include <hip/hip_runtime.h>
#include <math.h>

// ---------------------------------------------------------------------------
// MI455X (gfx1250) implementation: bf16 WMMA (V_WMMA_F32_16X16X32_BF16) for all
// GEMMs, f32 accumulate.  Wave tile = 32x32 (MT=2 x NT=2, 4 accumulators) so
// each weight fragment is loaded exactly once per step (M=32 recurrent GEMMs
// are L2-weight-bandwidth bound).
// ---------------------------------------------------------------------------

typedef __bf16 bf16_t;
typedef __bf16 v16bf __attribute__((ext_vector_type(16)));
typedef float  v8f   __attribute__((ext_vector_type(8)));

#define T_   32
#define B_   32
#define L_   128
#define F_   512
#define H_   1024
#define G4H  4096
#define SCALE_ 0.08838834764831845f  // 1/sqrt(128)

union Frag {
    float4 q[2];
    v16bf  v;
};

#define WMMA_BF16(A, Bm, C) \
    __builtin_amdgcn_wmma_f32_16x16x32_bf16(false, (A), false, (Bm), (short)0, (C), false, false)

// ---------------------------------------------------------------------------
// Generic bf16 GEMM:  out[M,N] = act( A[M,K] * W[N,K]^T + bias[N] )
// One wave computes a 32x32 output tile: 4 x 16x16 WMMA accumulators.
// Per K-step: 8 b128 loads -> 4 WMMAs (A shared across N, W shared across M).
// Wave-uniform guards only -> EXEC all ones at every WMMA.
// Requires M % 32 == 0, N % 32 == 0, K % 32 == 0.
// ---------------------------------------------------------------------------
__global__ void k_gemm_bf16(const bf16_t* __restrict__ A, int lda,
                            const bf16_t* __restrict__ W, int ldw,
                            const float* __restrict__ bias,
                            float* __restrict__ outF,
                            bf16_t* __restrict__ outB, int ldo,
                            int M, int N, int K, int relu)
{
    const int lane   = threadIdx.x & 31;
    const int wave   = blockIdx.x * (blockDim.x >> 5) + (threadIdx.x >> 5);
    const int tilesN = N >> 5;                 // 32-column tiles
    if (wave >= (M >> 5) * tilesN) return;     // uniform per wave
    const int tm = wave / tilesN;              // 32-row tile index
    const int tn = wave % tilesN;              // 32-col tile index
    const int r  = lane & 15;
    const int hi = lane >> 4;

    const bf16_t* aRow0 = A + (size_t)(tm * 32 + r) * lda;
    const bf16_t* aRow1 = A + (size_t)(tm * 32 + 16 + r) * lda;
    const bf16_t* bRow0 = W + (size_t)(tn * 32 + r) * ldw;
    const bf16_t* bRow1 = W + (size_t)(tn * 32 + 16 + r) * ldw;

    v8f acc00 = {};   // rows [0,16)  cols [0,16)
    v8f acc01 = {};   // rows [0,16)  cols [16,32)
    v8f acc10 = {};   // rows [16,32) cols [0,16)
    v8f acc11 = {};   // rows [16,32) cols [16,32)

    for (int kk = 0; kk < K; kk += 32) {
        Frag a0, a1, b0, b1;
        // A 16x32 bf16 layout: lanes 0-15 -> K[0..7],[16..23] ; 16-31 -> K[8..15],[24..31]
        a0.q[0] = *(const float4*)(aRow0 + kk + hi * 8);
        a0.q[1] = *(const float4*)(aRow0 + kk + 16 + hi * 8);
        a1.q[0] = *(const float4*)(aRow1 + kk + hi * 8);
        a1.q[1] = *(const float4*)(aRow1 + kk + 16 + hi * 8);
        // B 32x16 layout: lane = column; lanes 0-15 hold K[0..7],[16..23] ; 16-31 hold K[8..15],[24..31]
        const bf16_t* p0 = bRow0 + kk + hi * 8;
        b0.q[0] = *(const float4*)(p0);
        b0.q[1] = *(const float4*)(p0 + 16);
        const bf16_t* p1 = bRow1 + kk + hi * 8;
        b1.q[0] = *(const float4*)(p1);
        b1.q[1] = *(const float4*)(p1 + 16);

        acc00 = WMMA_BF16(a0.v, b0.v, acc00);
        acc01 = WMMA_BF16(a0.v, b1.v, acc01);
        acc10 = WMMA_BF16(a1.v, b0.v, acc10);
        acc11 = WMMA_BF16(a1.v, b1.v, acc11);
    }

    // C/D layout: VGPR i -> row = base + i + 8*hi, col = base + r (+16)
    const int   col0 = tn * 32 + r;
    const int   col1 = col0 + 16;
    const float bia0 = bias ? bias[col0] : 0.0f;
    const float bia1 = bias ? bias[col1] : 0.0f;
#pragma unroll
    for (int i = 0; i < 8; ++i) {
        const int row0 = tm * 32 + hi * 8 + i;
        const int row1 = row0 + 16;
        float v00 = acc00[i] + bia0;
        float v01 = acc01[i] + bia1;
        float v10 = acc10[i] + bia0;
        float v11 = acc11[i] + bia1;
        if (relu) {
            v00 = fmaxf(v00, 0.0f); v01 = fmaxf(v01, 0.0f);
            v10 = fmaxf(v10, 0.0f); v11 = fmaxf(v11, 0.0f);
        }
        if (outF) {
            outF[(size_t)row0 * ldo + col0] = v00;
            outF[(size_t)row0 * ldo + col1] = v01;
            outF[(size_t)row1 * ldo + col0] = v10;
            outF[(size_t)row1 * ldo + col1] = v11;
        }
        if (outB) {
            outB[(size_t)row0 * ldo + col0] = (bf16_t)v00;
            outB[(size_t)row0 * ldo + col1] = (bf16_t)v01;
            outB[(size_t)row1 * ldo + col0] = (bf16_t)v10;
            outB[(size_t)row1 * ldo + col1] = (bf16_t)v11;
        }
    }
}

// ---------------------------------------------------------------------------
// Elementwise helpers
// ---------------------------------------------------------------------------
__global__ void k_f32_to_bf16(bf16_t* __restrict__ dst, const float* __restrict__ src, int n)
{
    int i = blockIdx.x * blockDim.x + threadIdx.x;
    if (i < n) dst[i] = (bf16_t)src[i];
}

// dst[N, K1+K2] = [ src1[N,K1] | src2[N,K2] ]  (f32 -> bf16)
__global__ void k_concat_w(bf16_t* __restrict__ dst,
                           const float* __restrict__ src1, int K1,
                           const float* __restrict__ src2, int K2, int N)
{
    int i = blockIdx.x * blockDim.x + threadIdx.x;
    int Kt = K1 + K2;
    if (i >= N * Kt) return;
    int row = i / Kt, col = i % Kt;
    float v = (col < K1) ? src1[(size_t)row * K1 + col]
                         : src2[(size_t)row * K2 + (col - K1)];
    dst[i] = (bf16_t)v;
}

__global__ void k_add_bias(float* __restrict__ dst, const float* __restrict__ a,
                           const float* __restrict__ b, int n)
{
    int i = blockIdx.x * blockDim.x + threadIdx.x;
    if (i < n) dst[i] = a[i] + b[i];
}

// init: encoder h part of z = 0, cEnc = 0, decoder state from h0/c0
__global__ void k_init(bf16_t* __restrict__ zenc, float* __restrict__ cEnc,
                       bf16_t* __restrict__ zdec2, float* __restrict__ hDec,
                       float* __restrict__ cDec,
                       const float* __restrict__ h0, const float* __restrict__ c0)
{
    int idx = blockIdx.x * blockDim.x + threadIdx.x;
    if (idx >= B_ * H_) return;
    int b = idx >> 10, j = idx & (H_ - 1);
    zenc[(size_t)b * (F_ + H_) + F_ + j] = (bf16_t)0.0f;
    cEnc[idx] = 0.0f;
    float h = h0[idx], c = c0[idx];
    hDec[idx] = h;
    cDec[idx] = c;
    zdec2[(size_t)b * (2 * H_) + H_ + j] = (bf16_t)h;
}

// zenc[b, 0:F] = demo_bf16[b, l, :]
__global__ void k_pack_x_enc(bf16_t* __restrict__ zenc,
                             const bf16_t* __restrict__ demoBf, int l)
{
    int idx = blockIdx.x * blockDim.x + threadIdx.x;
    if (idx >= B_ * F_) return;
    int b = idx / F_, f = idx % F_;
    zenc[(size_t)b * (F_ + H_) + f] = demoBf[((size_t)b * L_ + l) * F_ + f];
}

__device__ __forceinline__ float sigm_(float x) { return 1.0f / (1.0f + __expf(-x)); }

// encoder LSTM pointwise: gates(B,4H f32) -> c, h; h feeds zenc and demoEnc(bf16)
__global__ void k_lstm_enc_pw(const float* __restrict__ gates,
                              float* __restrict__ cEnc,
                              bf16_t* __restrict__ zenc,
                              bf16_t* __restrict__ demoEncBf, int l)
{
    int idx = blockIdx.x * blockDim.x + threadIdx.x;
    if (idx >= B_ * H_) return;
    int b = idx >> 10, j = idx & (H_ - 1);
    const float* g = gates + (size_t)b * G4H;
    float ii = sigm_(g[j]);
    float ff = sigm_(g[H_ + j]);
    float gg = tanhf(g[2 * H_ + j]);
    float oo = sigm_(g[3 * H_ + j]);
    float c  = ff * cEnc[idx] + ii * gg;
    cEnc[idx] = c;
    float h = oo * tanhf(c);
    zenc[(size_t)b * (F_ + H_) + F_ + j]      = (bf16_t)h;
    demoEncBf[((size_t)l * B_ + b) * H_ + j]  = (bf16_t)h;
}

// decoder LSTM pointwise: also writes q[t] (f32 output) and h into zdec2 (bf16)
__global__ void k_lstm_dec_pw(const float* __restrict__ gates,
                              float* __restrict__ cDec, float* __restrict__ hDec,
                              bf16_t* __restrict__ zdec2, float* __restrict__ q, int t)
{
    int idx = blockIdx.x * blockDim.x + threadIdx.x;
    if (idx >= B_ * H_) return;
    int b = idx >> 10, j = idx & (H_ - 1);
    const float* g = gates + (size_t)b * G4H;
    float ii = sigm_(g[j]);
    float ff = sigm_(g[H_ + j]);
    float gg = tanhf(g[2 * H_ + j]);
    float oo = sigm_(g[3 * H_ + j]);
    float c  = ff * cDec[idx] + ii * gg;
    cDec[idx] = c;
    float h = oo * tanhf(c);
    hDec[idx] = h;
    q[(size_t)t * B_ * H_ + idx] = h;
    zdec2[(size_t)b * (2 * H_) + H_ + j] = (bf16_t)h;
}

// fused attention: scores -> masked softmax -> ctx ; also packs obs into z1
// one block per batch element
__global__ void k_attention(const bf16_t* __restrict__ attnProj,   // (L,B,H)
                            const bf16_t* __restrict__ demoEnc,    // (L,B,H)
                            const float* __restrict__ hDec,
                            const int* __restrict__ lengths,
                            const bf16_t* __restrict__ obsBf,      // (T,B,F)
                            bf16_t* __restrict__ z1, int t)        // (B, H+F)
{
    __shared__ float h_s[H_];
    __shared__ float w_s[L_];
    __shared__ float red[L_];
    const int b = blockIdx.x;
    const int tid = threadIdx.x;

    for (int j = tid; j < H_; j += blockDim.x) h_s[j] = hDec[b * H_ + j];
    __syncthreads();

    if (tid < L_) {
        const bf16_t* ap = attnProj + ((size_t)tid * B_ + b) * H_;
        float s = 0.0f;
        for (int j = 0; j < H_; ++j) s += (float)ap[j] * h_s[j];
        const int len = lengths[b];
        const bool valid = (tid < len) || (tid == 0);
        float sc = valid ? s * SCALE_ : -3.0e38f;
        w_s[tid] = sc;
        red[tid] = sc;
    }
    __syncthreads();
    for (int s = L_ / 2; s > 0; s >>= 1) {
        if (tid < s) red[tid] = fmaxf(red[tid], red[tid + s]);
        __syncthreads();
    }
    const float mx = red[0];
    __syncthreads();
    if (tid < L_) {
        float e = __expf(w_s[tid] - mx);
        w_s[tid] = e;
        red[tid] = e;
    }
    __syncthreads();
    for (int s = L_ / 2; s > 0; s >>= 1) {
        if (tid < s) red[tid] += red[tid + s];
        __syncthreads();
    }
    const float inv = 1.0f / red[0];
    __syncthreads();

    for (int j = tid; j < H_; j += blockDim.x) {
        float acc = 0.0f;
        const bf16_t* de = demoEnc + (size_t)b * H_ + j;
        for (int l = 0; l < L_; ++l) acc += w_s[l] * (float)de[(size_t)l * B_ * H_];
        z1[(size_t)b * (H_ + F_) + j] = (bf16_t)(acc * inv);
    }
    for (int f = tid; f < F_; f += blockDim.x)
        z1[(size_t)b * (H_ + F_) + H_ + f] = obsBf[((size_t)t * B_ + b) * F_ + f];
}

__global__ void k_finalize(float* __restrict__ out, const float* __restrict__ hDec,
                           const float* __restrict__ cDec)
{
    int idx = blockIdx.x * blockDim.x + threadIdx.x;
    if (idx >= B_ * H_) return;
    out[(size_t)T_ * B_ * H_ + idx]            = hDec[idx];
    out[(size_t)T_ * B_ * H_ + B_ * H_ + idx]  = cDec[idx];
}

// ---------------------------------------------------------------------------
// Host driver
// ---------------------------------------------------------------------------
extern "C" void kernel_launch(void* const* d_in, const int* in_sizes, int n_in,
                              void* d_out, int out_size, void* d_ws, size_t ws_size,
                              hipStream_t stream)
{
    const float* demo = (const float*)d_in[0];
    const int*   lens = (const int*)d_in[1];
    const float* obs  = (const float*)d_in[2];
    const float* h0   = (const float*)d_in[3];
    const float* c0   = (const float*)d_in[4];
    const float* eWih = (const float*)d_in[5];
    const float* eWhh = (const float*)d_in[6];
    const float* eBih = (const float*)d_in[7];
    const float* eBhh = (const float*)d_in[8];
    const float* aW   = (const float*)d_in[9];
    const float* aB   = (const float*)d_in[10];
    const float* m1W  = (const float*)d_in[11];
    const float* m1B  = (const float*)d_in[12];
    const float* m2W  = (const float*)d_in[13];
    const float* m2B  = (const float*)d_in[14];
    const float* lWih = (const float*)d_in[15];
    const float* lWhh = (const float*)d_in[16];
    const float* lBih = (const float*)d_in[17];
    const float* lBhh = (const float*)d_in[18];
    float* out = (float*)d_out;

    char* ws = (char*)d_ws;
    size_t off = 0;
    auto alloc = [&](size_t bytes) -> char* {
        char* p = ws + off;
        off = (off + bytes + 255) & ~(size_t)255;
        return p;
    };

    bf16_t* wEncCat   = (bf16_t*)alloc((size_t)G4H * (F_ + H_) * 2);  // [Wih|Whh]
    bf16_t* wAttn     = (bf16_t*)alloc((size_t)H_ * H_ * 2);
    bf16_t* wMlp1     = (bf16_t*)alloc((size_t)G4H * (F_ + H_) * 2);
    bf16_t* wMlp2     = (bf16_t*)alloc((size_t)H_ * G4H * 2);
    bf16_t* wLstmCat  = (bf16_t*)alloc((size_t)G4H * (2 * H_) * 2);   // [Wih|Whh]
    bf16_t* demoBf    = (bf16_t*)alloc((size_t)B_ * L_ * F_ * 2);
    bf16_t* obsBf     = (bf16_t*)alloc((size_t)T_ * B_ * F_ * 2);
    bf16_t* demoEncBf = (bf16_t*)alloc((size_t)L_ * B_ * H_ * 2);
    bf16_t* attnProjBf= (bf16_t*)alloc((size_t)L_ * B_ * H_ * 2);
    float*  bEnc      = (float*)alloc((size_t)G4H * 4);
    float*  bLstm     = (float*)alloc((size_t)G4H * 4);
    bf16_t* zenc      = (bf16_t*)alloc((size_t)B_ * (F_ + H_) * 2);   // [x|h]
    float*  cEnc      = (float*)alloc((size_t)B_ * H_ * 4);
    float*  gates     = (float*)alloc((size_t)B_ * G4H * 4);
    bf16_t* zdec1     = (bf16_t*)alloc((size_t)B_ * (H_ + F_) * 2);   // [ctx|obs]
    bf16_t* mbuf      = (bf16_t*)alloc((size_t)B_ * G4H * 2);
    bf16_t* zdec2     = (bf16_t*)alloc((size_t)B_ * (2 * H_) * 2);    // [x2|h]
    float*  cDec      = (float*)alloc((size_t)B_ * H_ * 4);
    float*  hDec      = (float*)alloc((size_t)B_ * H_ * 4);

    auto ew = [&](int n) { return dim3((n + 255) / 256); };
    auto gemm = [&](const bf16_t* A, int lda, const bf16_t* Wm, int ldw,
                    const float* bias, float* oF, bf16_t* oB, int ldo,
                    int M, int N, int K, int relu) {
        int waves  = (M / 32) * (N / 32);
        int blocks = (waves + 7) / 8;
        k_gemm_bf16<<<blocks, 256, 0, stream>>>(A, lda, Wm, ldw, bias, oF, oB, ldo,
                                                M, N, K, relu);
    };

    // ---- one-time (per launch) weight / input preparation -------------------
    {
        int n;
        n = B_ * L_ * F_;        k_f32_to_bf16<<<ew(n), 256, 0, stream>>>(demoBf, demo, n);  // demonstration[0]
        n = T_ * B_ * F_;        k_f32_to_bf16<<<ew(n), 256, 0, stream>>>(obsBf, obs, n);
        n = H_ * H_;             k_f32_to_bf16<<<ew(n), 256, 0, stream>>>(wAttn, aW, n);
        n = G4H * (F_ + H_);     k_f32_to_bf16<<<ew(n), 256, 0, stream>>>(wMlp1, m1W, n);
        n = H_ * G4H;            k_f32_to_bf16<<<ew(n), 256, 0, stream>>>(wMlp2, m2W, n);
        n = G4H * (F_ + H_);     k_concat_w<<<ew(n), 256, 0, stream>>>(wEncCat, eWih, F_, eWhh, H_, G4H);
        n = G4H * (2 * H_);      k_concat_w<<<ew(n), 256, 0, stream>>>(wLstmCat, lWih, H_, lWhh, H_, G4H);
        k_add_bias<<<ew(G4H), 256, 0, stream>>>(bEnc, eBih, eBhh, G4H);
        k_add_bias<<<ew(G4H), 256, 0, stream>>>(bLstm, lBih, lBhh, G4H);
        k_init<<<ew(B_ * H_), 256, 0, stream>>>(zenc, cEnc, zdec2, hDec, cDec, h0, c0);
    }

    // ---- encoder LSTM over L: gates = [x_l | h] @ [Wih|Whh]^T + b ----------
    for (int l = 0; l < L_; ++l) {
        k_pack_x_enc<<<ew(B_ * F_), 256, 0, stream>>>(zenc, demoBf, l);
        gemm(zenc, F_ + H_, wEncCat, F_ + H_, bEnc, gates, nullptr, G4H,
             B_, G4H, F_ + H_, 0);
        k_lstm_enc_pw<<<ew(B_ * H_), 256, 0, stream>>>(gates, cEnc, zenc, demoEncBf, l);
    }

    // ---- attn_proj = demo_enc @ attn_W^T + attn_b   (M=L*B=4096) -----------
    gemm(demoEncBf, H_, wAttn, H_, aB, nullptr, attnProjBf, H_,
         L_ * B_, H_, H_, 0);

    // ---- decoder loop over T ----------------------------------------------
    for (int t = 0; t < T_; ++t) {
        k_attention<<<B_, 256, 0, stream>>>(attnProjBf, demoEncBf, hDec, lens,
                                            obsBf, zdec1, t);
        // m = relu(z1 @ W1^T + b1)
        gemm(zdec1, H_ + F_, wMlp1, H_ + F_, m1B, nullptr, mbuf, G4H,
             B_, G4H, H_ + F_, 1);
        // x2 = relu(m @ W2^T + b2) -> zdec2[:, 0:H]
        gemm(mbuf, G4H, wMlp2, G4H, m2B, nullptr, zdec2, 2 * H_,
             B_, H_, G4H, 1);
        // gates = [x2 | h] @ [Wih|Whh]^T + b
        gemm(zdec2, 2 * H_, wLstmCat, 2 * H_, bLstm, gates, nullptr, G4H,
             B_, G4H, 2 * H_, 0);
        k_lstm_dec_pw<<<ew(B_ * H_), 256, 0, stream>>>(gates, cDec, hDec, zdec2, out, t);
    }

    k_finalize<<<ew(B_ * H_), 256, 0, stream>>>(out, hDec, cDec);
}
